// HeteroMLP_37211596653152
// MI455X (gfx1250) — matrix operand order repaired and checked by reading, not compiled, not run on hardware
//
#include <hip/hip_runtime.h>
#include <hip/hip_bf16.h>

// ---------------------------------------------------------------------------
// HeteroMLP on gfx1250: type-bucketed fused 3-layer MLP using split-bf16
// (hi+lo) WMMA: x*w ~= xh*wh + xh*wl + xl*wh  (fp32-grade accuracy, bf16 rate)
// ---------------------------------------------------------------------------

typedef __attribute__((ext_vector_type(16))) __bf16 v16bf;
typedef __attribute__((ext_vector_type(8)))  __bf16 v8bf;
typedef __attribute__((ext_vector_type(8)))  float  v8f;

#define ROWS    32          // rows (atoms) per workgroup
#define THREADS 256         // 8 waves of 32
#define IN_CH   128
#define H0      512
#define H1      512
#define OUTC    256
#define NTYPES  10

union V16U { v16bf v; v8bf h[2]; };

// A-matrix (16-bit, 16x32) per ISA: lane<16 holds row M=lane, K = kt*32+{0..7,16..23};
// lane>=16 holds K = kt*32+{8..15,24..31}. Row-major LDS [row][K] => two b128 loads.
__device__ __forceinline__ v16bf load_a_tile(const __bf16* __restrict__ A, int K,
                                             int rowbase, int lane, int kt) {
  const int row = rowbase + (lane & 15);
  const int k0  = kt * 32 + ((lane >> 4) << 3);
  V16U u;
  u.h[0] = *(const v8bf*)(A + row * K + k0);
  u.h[1] = *(const v8bf*)(A + row * K + k0 + 16);
  return u.v;
}

// B-matrix (16-bit, 32x16): lane<16 -> N=lane, K=kt*32+{0..7,16..23};
// lane>=16 -> N=lane-16, K=kt*32+{8..15,24..31}.  Weights stored [out][in] (K contiguous)
// => two b128 global loads per tile (L2-resident).
__device__ __forceinline__ v16bf load_b_tile(const __bf16* __restrict__ W, int K,
                                             int colbase, int lane, int kt) {
  const int col = colbase + (lane & 15);
  const int k0  = kt * 32 + ((lane >> 4) << 3);
  const __bf16* p = W + (size_t)col * K + k0;
  V16U u;
  u.h[0] = *(const v8bf*)(p);
  u.h[1] = *(const v8bf*)(p + 16);
  return u.v;
}

// One layer for this block's 32 rows: O = [relu](A @ W^T + b).
// Each of the 8 waves owns NOUT/8 output columns (whole wave => EXEC all-ones
// around every WMMA). acc0 = rows 0-15, acc1 = rows 16-31.
template <int K, int NOUT, bool RELU, bool LAST>
__device__ __forceinline__ void layer_gemm(
    const __bf16* __restrict__ Ah, const __bf16* __restrict__ Al,
    const __bf16* __restrict__ Wh, const __bf16* __restrict__ Wl,
    const float*  __restrict__ bias,
    __bf16* __restrict__ Oh, __bf16* __restrict__ Ol,
    float*  __restrict__ outg, const int* __restrict__ permBlk,
    int lane, int wave)
{
  constexpr int CPW = NOUT / 8;   // columns per wave
  constexpr int CT  = CPW / 16;   // 16-wide column tiles per wave
  for (int ct = 0; ct < CT; ++ct) {
    const int colbase = wave * CPW + ct * 16;
    v8f acc0 = {0.f, 0.f, 0.f, 0.f, 0.f, 0.f, 0.f, 0.f};
    v8f acc1 = {0.f, 0.f, 0.f, 0.f, 0.f, 0.f, 0.f, 0.f};
    for (int kt = 0; kt < K / 32; ++kt) {
      v16bf bh  = load_b_tile(Wh, K, colbase, lane, kt);
      v16bf bl  = load_b_tile(Wl, K, colbase, lane, kt);
      v16bf ah0 = load_a_tile(Ah, K, 0,  lane, kt);
      v16bf al0 = load_a_tile(Al, K, 0,  lane, kt);
      v16bf ah1 = load_a_tile(Ah, K, 16, lane, kt);
      v16bf al1 = load_a_tile(Al, K, 16, lane, kt);
      // Interleave acc0/acc1 chains so independent WMMAs cover D->C RAW slots.
      acc0 = __builtin_amdgcn_wmma_f32_16x16x32_bf16(false, ah0, false, bh, (short)0, acc0, false, false);
      acc1 = __builtin_amdgcn_wmma_f32_16x16x32_bf16(false, ah1, false, bh, (short)0, acc1, false, false);
      acc0 = __builtin_amdgcn_wmma_f32_16x16x32_bf16(false, ah0, false, bl, (short)0, acc0, false, false);
      acc1 = __builtin_amdgcn_wmma_f32_16x16x32_bf16(false, ah1, false, bl, (short)0, acc1, false, false);
      acc0 = __builtin_amdgcn_wmma_f32_16x16x32_bf16(false, al0, false, bh, (short)0, acc0, false, false);
      acc1 = __builtin_amdgcn_wmma_f32_16x16x32_bf16(false, al1, false, bh, (short)0, acc1, false, false);
    }
    // C/D layout (32-bit, 16x16): VGPR i, lanes0-15 -> M=i, lanes16-31 -> M=8+i; N = lane&15.
    const int   col = colbase + (lane & 15);
    const float bv  = bias[col];
#pragma unroll
    for (int i = 0; i < 8; ++i) {
      float v0 = acc0[i] + bv;
      float v1 = acc1[i] + bv;
      if (RELU) { v0 = fmaxf(v0, 0.f); v1 = fmaxf(v1, 0.f); }
      const int r0 = i + ((lane >> 4) << 3);
      const int r1 = r0 + 16;
      if (!LAST) {
        __bf16 h0 = (__bf16)v0; __bf16 l0 = (__bf16)(v0 - (float)h0);
        __bf16 h1 = (__bf16)v1; __bf16 l1 = (__bf16)(v1 - (float)h1);
        Oh[r0 * NOUT + col] = h0;  Ol[r0 * NOUT + col] = l0;
        Oh[r1 * NOUT + col] = h1;  Ol[r1 * NOUT + col] = l1;
      } else {
        const int g0 = permBlk[r0];
        const int g1 = permBlk[r1];
        if (g0 >= 0) outg[(size_t)g0 * NOUT + col] = v0;
        if (g1 >= 0) outg[(size_t)g1 * NOUT + col] = v1;
      }
    }
  }
}

// --------------------------- bucketing kernels -----------------------------

__global__ void init_meta_kernel(int* counts, int* perm, int padcap) {
  const int i = blockIdx.x * blockDim.x + threadIdx.x;
  if (i < 16) counts[i] = 0;
  if (i < padcap) perm[i] = -1;
}

__global__ void histo_kernel(const int* __restrict__ types, int* counts, int n) {
  const int i = blockIdx.x * blockDim.x + threadIdx.x;
  if (i < n) atomicAdd(&counts[types[i]], 1);
}

__global__ void scan_kernel(const int* __restrict__ counts, int* cursors, int* poffs) {
  if (threadIdx.x == 0 && blockIdx.x == 0) {
    int acc = 0;
    poffs[0] = 0;
    for (int t = 0; t < NTYPES; ++t) {
      cursors[t] = acc;                                // segment start
      acc += ((counts[t] + (ROWS - 1)) / ROWS) * ROWS; // pad each type to ROWS
      poffs[t + 1] = acc;
    }
  }
}

__global__ void scatter_kernel(const int* __restrict__ types, int* cursors,
                               int* perm, int n) {
  const int i = blockIdx.x * blockDim.x + threadIdx.x;
  if (i < n) {
    const int t   = types[i];
    const int pos = atomicAdd(&cursors[t], 1);
    perm[pos] = i;
  }
}

// Split fp32 W[t][in][out] -> bf16 hi/lo W[t][out][in] (B-operand friendly).
__global__ void convert_w_kernel(const float* __restrict__ W0,
                                 const float* __restrict__ W1,
                                 const float* __restrict__ W2,
                                 __bf16* Wh0, __bf16* Wl0,
                                 __bf16* Wh1, __bf16* Wl1,
                                 __bf16* Wh2, __bf16* Wl2) {
  const int S0 = NTYPES * IN_CH * H0;
  const int S1 = NTYPES * H0 * H1;
  const int S2 = NTYPES * H1 * OUTC;
  const int idx = blockIdx.x * blockDim.x + threadIdx.x;
  if (idx < S0) {
    const int t = idx / (IN_CH * H0), rem = idx % (IN_CH * H0);
    const int k = rem / H0, nn = rem % H0;
    const float w = W0[idx];
    const __bf16 h = (__bf16)w;
    const int d = (t * H0 + nn) * IN_CH + k;
    Wh0[d] = h; Wl0[d] = (__bf16)(w - (float)h);
  } else if (idx < S0 + S1) {
    const int j = idx - S0;
    const int t = j / (H0 * H1), rem = j % (H0 * H1);
    const int k = rem / H1, nn = rem % H1;
    const float w = W1[j];
    const __bf16 h = (__bf16)w;
    const int d = (t * H1 + nn) * H0 + k;
    Wh1[d] = h; Wl1[d] = (__bf16)(w - (float)h);
  } else if (idx < S0 + S1 + S2) {
    const int j = idx - S0 - S1;
    const int t = j / (H1 * OUTC), rem = j % (H1 * OUTC);
    const int k = rem / OUTC, nn = rem % OUTC;
    const float w = W2[j];
    const __bf16 h = (__bf16)w;
    const int d = (t * OUTC + nn) * H1 + k;
    Wh2[d] = h; Wl2[d] = (__bf16)(w - (float)h);
  }
}

// --------------------------- fused MLP kernel ------------------------------

__global__ void __launch_bounds__(THREADS)
fused_mlp_kernel(const float* __restrict__ x,
                 const int* __restrict__ perm, const int* __restrict__ poffs,
                 const __bf16* __restrict__ Wh0, const __bf16* __restrict__ Wl0,
                 const float* __restrict__ b0,
                 const __bf16* __restrict__ Wh1, const __bf16* __restrict__ Wl1,
                 const float* __restrict__ b1,
                 const __bf16* __restrict__ Wh2, const __bf16* __restrict__ Wl2,
                 const float* __restrict__ b2,
                 float* __restrict__ out) {
  extern __shared__ char smem[];
  __bf16* Gh = (__bf16*)smem;        // [ROWS][512] ping (layer0 in / layer1 out)
  __bf16* Gl = Gh + ROWS * 512;
  __bf16* Hh = Gl + ROWS * 512;      // [ROWS][512] pong (layer0 out / layer1 in)
  __bf16* Hl = Hh + ROWS * 512;

  const int tid  = threadIdx.x;
  const int lane = tid & 31;
  const int wave = tid >> 5;
  const int base = blockIdx.x * ROWS;

  // Uniform per-block type lookup (padded segments => whole block is one type).
  int t = NTYPES - 1;
  for (int i = 0; i < NTYPES; ++i) {
    if (base < poffs[i + 1]) { t = i; break; }
  }
  const int* permBlk = perm + base;

  // Stage X (gathered via perm) into LDS as split bf16; sentinel rows -> 0.
  for (int e = tid; e < ROWS * IN_CH; e += THREADS) {
    const int r = e >> 7, c = e & (IN_CH - 1);
    const int g = permBlk[r];
    const float v = (g >= 0) ? x[(size_t)g * IN_CH + c] : 0.0f;
    const __bf16 h = (__bf16)v;
    Gh[r * IN_CH + c] = h;
    Gl[r * IN_CH + c] = (__bf16)(v - (float)h);
  }
  __syncthreads();

  layer_gemm<IN_CH, H0, true, false>(Gh, Gl,
      Wh0 + (size_t)t * H0 * IN_CH, Wl0 + (size_t)t * H0 * IN_CH,
      b0 + t * H0, Hh, Hl, nullptr, nullptr, lane, wave);
  __syncthreads();

  layer_gemm<H0, H1, true, false>(Hh, Hl,
      Wh1 + (size_t)t * H1 * H0, Wl1 + (size_t)t * H1 * H0,
      b1 + t * H1, Gh, Gl, nullptr, nullptr, lane, wave);
  __syncthreads();

  layer_gemm<H1, OUTC, false, true>(Gh, Gl,
      Wh2 + (size_t)t * OUTC * H1, Wl2 + (size_t)t * OUTC * H1,
      b2 + t * OUTC, nullptr, nullptr, out, permBlk, lane, wave);
}

// ------------------------------- launcher ----------------------------------

extern "C" void kernel_launch(void* const* d_in, const int* in_sizes, int n_in,
                              void* d_out, int out_size, void* d_ws, size_t ws_size,
                              hipStream_t stream) {
  const float* x     = (const float*)d_in[0];
  const int*   types = (const int*)d_in[1];
  const float* W0    = (const float*)d_in[2];
  const float* b0    = (const float*)d_in[3];
  const float* W1    = (const float*)d_in[4];
  const float* b1    = (const float*)d_in[5];
  const float* W2    = (const float*)d_in[6];
  const float* b2    = (const float*)d_in[7];
  float* out = (float*)d_out;

  const int N = in_sizes[1];                         // one type per atom
  const int PADCAP = ((N + ROWS - 1) / ROWS) * ROWS + NTYPES * ROWS;

  // Workspace layout (~19.2 MB total; assumes ws_size is at least that).
  char* ws = (char*)d_ws;
  int* counts  = (int*)(ws + 0);
  int* cursors = (int*)(ws + 64);
  int* poffs   = (int*)(ws + 128);
  int* perm    = (int*)(ws + 256);
  size_t woff = 256 + (size_t)PADCAP * sizeof(int);
  woff = (woff + 255) & ~(size_t)255;
  const size_t S0 = (size_t)NTYPES * IN_CH * H0;
  const size_t S1 = (size_t)NTYPES * H0 * H1;
  const size_t S2 = (size_t)NTYPES * H1 * OUTC;
  __bf16* Wh0 = (__bf16*)(ws + woff);
  __bf16* Wl0 = Wh0 + S0;
  __bf16* Wh1 = Wl0 + S0;
  __bf16* Wl1 = Wh1 + S1;
  __bf16* Wh2 = Wl1 + S1;
  __bf16* Wl2 = Wh2 + S2;

  init_meta_kernel<<<(PADCAP + 255) / 256, 256, 0, stream>>>(counts, perm, PADCAP);
  histo_kernel<<<(N + 255) / 256, 256, 0, stream>>>(types, counts, N);
  scan_kernel<<<1, 32, 0, stream>>>(counts, cursors, poffs);
  scatter_kernel<<<(N + 255) / 256, 256, 0, stream>>>(types, cursors, perm, N);

  const int ETOT = (int)(S0 + S1 + S2);
  convert_w_kernel<<<(ETOT + 255) / 256, 256, 0, stream>>>(
      W0, W1, W2, Wh0, Wl0, Wh1, Wl1, Wh2, Wl2);

  const int SMEM = 4 * ROWS * 512 * (int)sizeof(__bf16);  // 128 KB dynamic LDS
  (void)hipFuncSetAttribute((const void*)fused_mlp_kernel,
                            hipFuncAttributeMaxDynamicSharedMemorySize, SMEM);
  fused_mlp_kernel<<<PADCAP / ROWS, THREADS, SMEM, stream>>>(
      x, perm, poffs, Wh0, Wl0, b0, Wh1, Wl1, b1, Wh2, Wl2, b2, out);
}
